// MessagePassing_76733885710816
// MI455X (gfx1250) — matrix operand order repaired and checked by reading, not compiled, not run on hardware
//
#include <hip/hip_runtime.h>
#include <math.h>

#define NN 10000          // nodes
#define NE 40000          // edges
#define NG 128            // graphs
#define DD 48             // D = ATOM + STATE
#define KPR 1600          // padded fused-K rows for message GEMM

typedef __attribute__((ext_vector_type(16))) _Float16 v16h;
typedef __attribute__((ext_vector_type(8)))  float    v8f;
typedef __attribute__((ext_vector_type(4)))  unsigned int v4u;
typedef __attribute__((ext_vector_type(4)))  int      v4i;
typedef __attribute__((ext_vector_type(8)))  int      v8i;

__device__ __forceinline__ float sigm(float x) { return 1.0f / (1.0f + __expf(-x)); }

// ---------------------------------------------------------------------------
// Tensor Data Mover: prime LDS with a 2D f16 panel (rows x dim0 halves).
// D# built per CDNA5 ISA §8 (group0: count/lds/global/type=2; group1:
// data_size=2B, tensor/tile dims, stride).  Used as a warm-up DMA; a
// barrier-ordered plain cooperative load follows to guarantee contents.
// ---------------------------------------------------------------------------
__device__ __forceinline__ void tdm_load_halves(void* lds_dst, const _Float16* gsrc,
                                                unsigned dim0, unsigned rows) {
    unsigned lofs = (unsigned)(size_t)lds_dst;
    unsigned long long ga = (unsigned long long)(size_t)gsrc;
    v4u g0;
    g0[0] = 1u;                                               // count = 1 (valid user D#)
    g0[1] = lofs;                                             // lds_addr
    g0[2] = (unsigned)ga;                                     // global_addr[31:0]
    g0[3] = (unsigned)((ga >> 32) & 0x01FFFFFFu) | (2u << 30); // global_addr[56:32] | type=2
    v8i g1;
    g1[0] = 0x00010000;                                       // data_size = 2 bytes
    g1[1] = (int)((dim0 & 0xFFFFu) << 16);                    // tensor_dim0[15:0]
    g1[2] = (int)(((dim0 >> 16) & 0xFFFFu) | ((rows & 0xFFFFu) << 16)); // dim0 hi | dim1 lo
    g1[3] = (int)(((rows >> 16) & 0xFFFFu) | ((dim0 & 0xFFFFu) << 16)); // dim1 hi | tile_dim0
    g1[4] = (int)(rows & 0xFFFFu);                            // tile_dim1 | tile_dim2=0
    g1[5] = (int)dim0;                                        // tensor_dim0_stride lo
    g1[6] = 0;
    g1[7] = 0;
    v4i gz = {0, 0, 0, 0};
#if defined(__clang_major__) && (__clang_major__ >= 23)
    v8i gz8 = {0, 0, 0, 0, 0, 0, 0, 0};
    __builtin_amdgcn_tensor_load_to_lds(g0, g1, gz, gz, gz8, 0);
#else
    __builtin_amdgcn_tensor_load_to_lds(g0, g1, gz, gz, 0);
#endif
}

// ---------------------------------------------------------------------------
// Pack kernel (32 x 2304) + bias into K'' (48 cols x 1600 rows, col-major f16)
// r = j*32 + k (r<1536); [1536,1584): bias rows; rest zero.
// ---------------------------------------------------------------------------
__global__ void prep_kp(const float* __restrict__ kmat, const float* __restrict__ bias,
                        _Float16* __restrict__ Kp) {
    int idx = blockIdx.x * blockDim.x + threadIdx.x;
    if (idx >= DD * KPR) return;
    int col = idx / KPR;
    int r   = idx - col * KPR;
    float v = 0.0f;
    if (r < 1536) {
        int k = r & 31;
        int j = r >> 5;
        v = kmat[k * (DD * DD) + col * DD + j];
    } else if (r < 1536 + DD) {
        v = bias[col * DD + (r - 1536)];
    }
    Kp[idx] = (_Float16)v;
}

// ---------------------------------------------------------------------------
// Pack edge-GRU weights: Wc (128 cols x 160 rows, col-major f16).
// Rows 0..111 = We, 112..143 = Ue, rest 0.  Col groups (32 each):
// 0: z-cols, 1: r-cols, 2: xh (We only), 3: hh (Ue only).
// ---------------------------------------------------------------------------
__global__ void prep_wc(const float* __restrict__ We, const float* __restrict__ Ue,
                        _Float16* __restrict__ Wc) {
    int idx = blockIdx.x * blockDim.x + threadIdx.x;
    if (idx >= 128 * 160) return;
    int col = idx / 160, r = idx - col * 160;
    int g = col >> 5, j = col & 31;
    float v = 0.0f;
    if (r < 112) {
        if (g == 0)      v = We[r * 96 + j];
        else if (g == 1) v = We[r * 96 + 32 + j];
        else if (g == 2) v = We[r * 96 + 64 + j];
    } else if (r < 144) {
        int ru = r - 112;
        if (g == 0)      v = Ue[ru * 96 + j];
        else if (g == 1) v = Ue[ru * 96 + 32 + j];
        else if (g == 3) v = Ue[ru * 96 + 64 + j];
    }
    Wc[idx] = (_Float16)v;
}

// amc[n] = concat(a[n], s[agi[n]])
__global__ void build_amc(const float* __restrict__ a, const float* __restrict__ s,
                          const int* __restrict__ agi, float* __restrict__ amc) {
    int i = blockIdx.x * blockDim.x + threadIdx.x;
    if (i >= NN * DD) return;
    int n = i / DD, j = i - n * DD;
    amc[i] = (j < 32) ? a[n * 32 + j] : s[agi[n] * 16 + (j - 32)];
}

// ---------------------------------------------------------------------------
// Dominant GEMM: t = Z @ K'' with 32-edge tiles per wave (two A sub-tiles
// share every B fetch -> 6 WMMA per 6 ds_load_b128).
// ---------------------------------------------------------------------------
#define MSG_WAVES 8
#define MSG_BLOCKS 80
__global__ __launch_bounds__(256) void msg_wmma(
    const float* __restrict__ e,      // NE x 32 (post edge-GRU)
    const float* __restrict__ amc,    // NN x 48
    const int*   __restrict__ pair,   // NE x 2 (src,dst)
    const _Float16* __restrict__ Kp,  // 48 x 1600 col-major f16
    float* __restrict__ agg)          // NN x 48 (pre-zeroed, atomic)
{
    __shared__ __align__(32) _Float16 Bs[DD * KPR];           // 150 KB
    __shared__ __align__(16) float    Am[MSG_WAVES][32 * 64]; // 64 KB

    const int tid = threadIdx.x;
    if (tid < 32) {                       // wave 0: TDM warm-up DMA
        tdm_load_halves(Bs, Kp, 4800u, 16u);
        __builtin_amdgcn_s_wait_tensorcnt(0);
    }
    __syncthreads();
    for (int i = tid; i < DD * KPR / 8; i += 256)
        ((uint4*)Bs)[i] = ((const uint4*)Kp)[i];
    __syncthreads();

    const int wave  = tid >> 5;
    const int lane  = tid & 31;
    const int hi    = lane >> 4;
    const int m     = lane & 15;
    const int kofs  = hi ? 16 : 0;
    const int khalf = hi << 4;
    const int nTiles = NE / 32;           // 1250

    for (int tile = blockIdx.x * MSG_WAVES + wave; tile < nTiles;
         tile += gridDim.x * MSG_WAVES) {
        const int ebase = tile * 32;
        float* amrow = &Am[wave][0];

        int nxt = tile + gridDim.x * MSG_WAVES;
        if (nxt < nTiles) {
            __builtin_prefetch(&e[nxt * 32 * 32], 0, 3);
            __builtin_prefetch(&pair[nxt * 64], 0, 3);
        }

        // Stage 32 amc rows (gathered via dst) as float4, zero-padded to 64
        for (int i = lane; i < 32 * 16; i += 32) {
            int mm = i >> 4, q = i & 15;
            int j = q * 4;
            int d = pair[(ebase + mm) * 2 + 1];
            float4 v = (j < DD) ? *(const float4*)&amc[d * DD + j]
                                : make_float4(0.f, 0.f, 0.f, 0.f);
            *(float4*)&amrow[mm * 64 + j] = v;
        }

        // Two packed f16 e-rows in WMMA-A order
        const float* ep0 = e + (ebase + m) * 32 + kofs;
        const float* ep1 = e + (ebase + 16 + m) * 32 + kofs;
        v16h eh0, eh1;
        #pragma unroll
        for (int i = 0; i < 8; ++i) {
            eh0[i] = (_Float16)ep0[i]; eh0[i + 8] = (_Float16)ep0[i + 8];
            eh1[i] = (_Float16)ep1[i]; eh1[i + 8] = (_Float16)ep1[i + 8];
        }

        v8f acc[6] = {};
        const float* mr0 = &amrow[m * 64];
        const float* mr1 = &amrow[(16 + m) * 64];
        const _Float16* bb = &Bs[m * KPR + khalf];

        #pragma unroll 4
        for (int c = 0; c < 48; ++c) {
            _Float16 s0 = (_Float16)mr0[c];
            _Float16 s1 = (_Float16)mr1[c];
            v16h a0 = eh0 * s0;
            v16h a1 = eh1 * s1;
            const _Float16* bc = bb + c * 32;
            v16h b0 = *(const v16h*)(bc);
            v16h b1 = *(const v16h*)(bc + 16 * KPR);
            v16h b2 = *(const v16h*)(bc + 32 * KPR);
            acc[0] = __builtin_amdgcn_wmma_f32_16x16x32_f16(false, a0, false, b0, (short)0, acc[0], false, false);
            acc[1] = __builtin_amdgcn_wmma_f32_16x16x32_f16(false, a0, false, b1, (short)0, acc[1], false, false);
            acc[2] = __builtin_amdgcn_wmma_f32_16x16x32_f16(false, a0, false, b2, (short)0, acc[2], false, false);
            acc[3] = __builtin_amdgcn_wmma_f32_16x16x32_f16(false, a1, false, b0, (short)0, acc[3], false, false);
            acc[4] = __builtin_amdgcn_wmma_f32_16x16x32_f16(false, a1, false, b1, (short)0, acc[4], false, false);
            acc[5] = __builtin_amdgcn_wmma_f32_16x16x32_f16(false, a1, false, b2, (short)0, acc[5], false, false);
        }
        // Tail chunks 48,49: A = amc values (bias term), zeros beyond 48
        #pragma unroll
        for (int c = 48; c < 50; ++c) {
            const float* p0 = &mr0[(c - 48) * 32 + kofs];
            const float* p1 = &mr1[(c - 48) * 32 + kofs];
            v16h a0, a1;
            #pragma unroll
            for (int i = 0; i < 8; ++i) {
                a0[i] = (_Float16)p0[i]; a0[i + 8] = (_Float16)p0[i + 8];
                a1[i] = (_Float16)p1[i]; a1[i + 8] = (_Float16)p1[i + 8];
            }
            const _Float16* bc = bb + c * 32;
            v16h b0 = *(const v16h*)(bc);
            v16h b1 = *(const v16h*)(bc + 16 * KPR);
            v16h b2 = *(const v16h*)(bc + 32 * KPR);
            acc[0] = __builtin_amdgcn_wmma_f32_16x16x32_f16(false, a0, false, b0, (short)0, acc[0], false, false);
            acc[1] = __builtin_amdgcn_wmma_f32_16x16x32_f16(false, a0, false, b1, (short)0, acc[1], false, false);
            acc[2] = __builtin_amdgcn_wmma_f32_16x16x32_f16(false, a0, false, b2, (short)0, acc[2], false, false);
            acc[3] = __builtin_amdgcn_wmma_f32_16x16x32_f16(false, a1, false, b0, (short)0, acc[3], false, false);
            acc[4] = __builtin_amdgcn_wmma_f32_16x16x32_f16(false, a1, false, b1, (short)0, acc[4], false, false);
            acc[5] = __builtin_amdgcn_wmma_f32_16x16x32_f16(false, a1, false, b2, (short)0, acc[5], false, false);
        }

        // Scatter into agg[src]
        #pragma unroll
        for (int v = 0; v < 8; ++v) {
            int r0 = ebase + v + hi * 8;
            int r1 = r0 + 16;
            int s0i = pair[r0 * 2];
            int s1i = pair[r1 * 2];
            float* dp0 = &agg[s0i * DD + m];
            float* dp1 = &agg[s1i * DD + m];
            atomicAdd(dp0 + 0,  acc[0][v]);
            atomicAdd(dp0 + 16, acc[1][v]);
            atomicAdd(dp0 + 32, acc[2][v]);
            atomicAdd(dp1 + 0,  acc[3][v]);
            atomicAdd(dp1 + 16, acc[4][v]);
            atomicAdd(dp1 + 32, acc[5][v]);
        }
    }
}

// ---------------------------------------------------------------------------
// Edge GRU as WMMA GEMM: [x|h] (16 x 160) @ Wc (160 x 128), then per-lane
// GRU nonlinearity epilogue (z/r/xh/hh for the same (row,j) land in the
// same lane's accumulators).
// ---------------------------------------------------------------------------
#define EG_WAVES 8
#define EG_BLOCKS 80
__global__ __launch_bounds__(256) void edge_gru_wmma(
    const float* __restrict__ a, const float* __restrict__ e_in,
    const float* __restrict__ s, const int* __restrict__ pair,
    const int* __restrict__ bgi, const _Float16* __restrict__ Wc,
    const float* __restrict__ bi, const float* __restrict__ br,
    float* __restrict__ e_out)
{
    __shared__ __align__(32) _Float16 Wcs[128 * 160];         // 40 KB
    __shared__ __align__(16) float    Xs[EG_WAVES][16 * 160]; // 80 KB

    const int tid = threadIdx.x;
    if (tid < 32) {
        tdm_load_halves(Wcs, Wc, 1280u, 16u);
        __builtin_amdgcn_s_wait_tensorcnt(0);
    }
    __syncthreads();
    for (int i = tid; i < 128 * 160 / 8; i += 256)
        ((uint4*)Wcs)[i] = ((const uint4*)Wc)[i];
    __syncthreads();

    const int wave  = tid >> 5;
    const int lane  = tid & 31;
    const int hi    = lane >> 4;
    const int m     = lane & 15;
    const int kofs  = hi ? 16 : 0;
    const int khalf = hi << 4;
    const int nTiles = NE / 16;           // 2500

    for (int tile = blockIdx.x * EG_WAVES + wave; tile < nTiles;
         tile += gridDim.x * EG_WAVES) {
        const int ebase = tile * 16;
        float* xrow = &Xs[wave][0];

        // Stage X rows as float4 (all segment boundaries are multiples of 4):
        // [a[src](32) | a[dst](32) | s[bgi](16) | e(32) | e(32, h-copy) | 0(16)]
        for (int i = lane; i < 16 * 40; i += 32) {
            int mm = i / 40, q = i - mm * 40;
            int k = q * 4;
            int ei = ebase + mm;
            float4 v;
            if (k < 32)       v = *(const float4*)&a[pair[ei * 2] * 32 + k];
            else if (k < 64)  v = *(const float4*)&a[pair[ei * 2 + 1] * 32 + (k - 32)];
            else if (k < 80)  v = *(const float4*)&s[bgi[ei] * 16 + (k - 64)];
            else if (k < 112) v = *(const float4*)&e_in[ei * 32 + (k - 80)];
            else if (k < 144) v = *(const float4*)&e_in[ei * 32 + (k - 112)];
            else              v = make_float4(0.f, 0.f, 0.f, 0.f);
            *(float4*)&xrow[mm * 160 + k] = v;
        }

        const float* mr = &xrow[m * 160];
        v8f acc[8] = {};
        #pragma unroll
        for (int c = 0; c < 5; ++c) {
            const float* p0 = &mr[c * 32 + kofs];
            v16h aT;
            #pragma unroll
            for (int i = 0; i < 8; ++i) {
                aT[i] = (_Float16)p0[i]; aT[i + 8] = (_Float16)p0[i + 8];
            }
            #pragma unroll
            for (int t = 0; t < 8; ++t) {
                const _Float16* bc = &Wcs[(t * 16 + m) * 160 + c * 32 + khalf];
                v16h b = *(const v16h*)bc;
                acc[t] = __builtin_amdgcn_wmma_f32_16x16x32_f16(false, aT, false, b, (short)0, acc[t], false, false);
            }
        }

        // GRU nonlinearity epilogue
        #pragma unroll
        for (int v = 0; v < 8; ++v) {
            int row = ebase + v + hi * 8;
            #pragma unroll
            for (int jj = 0; jj < 2; ++jj) {
                int j = m + jj * 16;
                float z  = sigm(acc[0 + jj][v] + bi[j] + br[j]);
                float r  = sigm(acc[2 + jj][v] + bi[32 + j] + br[32 + j]);
                float hc = tanhf(acc[4 + jj][v] + bi[64 + j] +
                                 r * (acc[6 + jj][v] + br[64 + j]));
                e_out[row * 32 + j] = z * e_in[row * 32 + j] + (1.0f - z) * hc;
            }
        }
    }
}

// ---------------------------------------------------------------------------
// Node GRU (small: 0.15 GFLOP): LDS-weight VALU kernel
// ---------------------------------------------------------------------------
__global__ __launch_bounds__(256) void node_gru(
    const float* __restrict__ agg, const float* __restrict__ a_in,
    const float* __restrict__ Wn, const float* __restrict__ Un,
    const float* __restrict__ b_in, const float* __restrict__ b_rec,
    float* __restrict__ a_out)
{
    __shared__ float Wl[48 * 96];
    __shared__ float Ul[32 * 96];
    __shared__ float bi[96], br[96];
    for (int i = threadIdx.x; i < 48 * 96; i += 256) Wl[i] = Wn[i];
    for (int i = threadIdx.x; i < 32 * 96; i += 256) Ul[i] = Un[i];
    for (int i = threadIdx.x; i < 96; i += 256) { bi[i] = b_in[i]; br[i] = b_rec[i]; }
    __syncthreads();
    int n = blockIdx.x * 256 + threadIdx.x;
    if (n >= NN) return;
    float x[48], h[32];
    #pragma unroll
    for (int k = 0; k < 48; ++k) x[k] = agg[n * 48 + k];
    #pragma unroll
    for (int k = 0; k < 32; ++k) h[k] = a_in[n * 32 + k];
    for (int j = 0; j < 32; ++j) {
        float xz = bi[j], xr = bi[j + 32], xh = bi[j + 64];
        float hz = br[j], hr = br[j + 32], hh = br[j + 64];
        #pragma unroll
        for (int k = 0; k < 48; ++k) {
            float v = x[k]; const float* w = &Wl[k * 96];
            xz += v * w[j]; xr += v * w[j + 32]; xh += v * w[j + 64];
        }
        #pragma unroll
        for (int k = 0; k < 32; ++k) {
            float v = h[k]; const float* u = &Ul[k * 96];
            hz += v * u[j]; hr += v * u[j + 32]; hh += v * u[j + 64];
        }
        float z = sigm(xz + hz), r = sigm(xr + hr);
        float hc = tanhf(xh + r * hh);
        a_out[n * 32 + j] = z * a_in[n * 32 + j] + (1.0f - z) * hc;
    }
}

// segment-sum of 32-wide rows into out[seg[n]]
__global__ void seg_sum32(const float* __restrict__ x, const int* __restrict__ seg,
                          float* __restrict__ out, int rows) {
    int i = blockIdx.x * blockDim.x + threadIdx.x;
    if (i >= rows * 32) return;
    int n = i >> 5, j = i & 31;
    atomicAdd(&out[seg[n] * 32 + j], x[i]);
}

// ---------------------------------------------------------------------------
// State GRU (tiny): x = [asum(32), esum(32), s(16)]
// ---------------------------------------------------------------------------
__global__ __launch_bounds__(128) void state_gru(
    const float* __restrict__ asum, const float* __restrict__ esum,
    const float* __restrict__ s_in,
    const float* __restrict__ Wsx, const float* __restrict__ Usx,
    const float* __restrict__ b_in, const float* __restrict__ b_rec,
    float* __restrict__ s_out)
{
    __shared__ float Wl[80 * 48];
    __shared__ float Ul[16 * 48];
    for (int i = threadIdx.x; i < 80 * 48; i += 128) Wl[i] = Wsx[i];
    for (int i = threadIdx.x; i < 16 * 48; i += 128) Ul[i] = Usx[i];
    __syncthreads();
    int g = threadIdx.x;
    if (g >= NG) return;
    float x[80];
    #pragma unroll
    for (int k = 0; k < 32; ++k) x[k]      = asum[g * 32 + k];
    #pragma unroll
    for (int k = 0; k < 32; ++k) x[32 + k] = esum[g * 32 + k];
    #pragma unroll
    for (int k = 0; k < 16; ++k) x[64 + k] = s_in[g * 16 + k];
    for (int j = 0; j < 16; ++j) {
        float xz = b_in[j], xr = b_in[j + 16], xh = b_in[j + 32];
        float hz = b_rec[j], hr = b_rec[j + 16], hh = b_rec[j + 32];
        #pragma unroll
        for (int k = 0; k < 80; ++k) {
            float v = x[k]; const float* w = &Wl[k * 48];
            xz += v * w[j]; xr += v * w[j + 16]; xh += v * w[j + 32];
        }
        #pragma unroll
        for (int k = 0; k < 16; ++k) {
            float v = x[64 + k]; const float* u = &Ul[k * 48];
            hz += v * u[j]; hr += v * u[j + 16]; hh += v * u[j + 32];
        }
        float z = sigm(xz + hz), r = sigm(xr + hr);
        float hc = tanhf(xh + r * hh);
        s_out[g * 16 + j] = z * s_in[g * 16 + j] + (1.0f - z) * hc;
    }
}

// ---------------------------------------------------------------------------
extern "C" void kernel_launch(void* const* d_in, const int* in_sizes, int n_in,
                              void* d_out, int out_size, void* d_ws, size_t ws_size,
                              hipStream_t stream)
{
    const float* a0   = (const float*)d_in[0];
    const float* e0   = (const float*)d_in[1];
    const float* s0   = (const float*)d_in[2];
    const int*   pair = (const int*)d_in[3];
    const int*   agi  = (const int*)d_in[4];
    const int*   bgi  = (const int*)d_in[5];
    const float* kmat = (const float*)d_in[6];
    const float* bias = (const float*)d_in[7];
    const float* We   = (const float*)d_in[8];
    const float* Ue   = (const float*)d_in[9];
    const float* bei  = (const float*)d_in[10];
    const float* ber  = (const float*)d_in[11];
    const float* Wn   = (const float*)d_in[12];
    const float* Un   = (const float*)d_in[13];
    const float* bni  = (const float*)d_in[14];
    const float* bnr  = (const float*)d_in[15];
    const float* Wsx  = (const float*)d_in[16];
    const float* Usx  = (const float*)d_in[17];
    const float* bsi  = (const float*)d_in[18];
    const float* bsr  = (const float*)d_in[19];

    char* ws = (char*)d_ws;
    auto alloc = [&](size_t bytes) -> char* {
        char* p = ws; ws += (bytes + 255) & ~size_t(255); return p;
    };
    _Float16* Kp  = (_Float16*)alloc(DD * KPR * sizeof(_Float16));
    _Float16* Wc  = (_Float16*)alloc(128 * 160 * sizeof(_Float16));
    float* aB[2]  = {(float*)alloc(NN * 32 * 4), (float*)alloc(NN * 32 * 4)};
    float* eB[2]  = {(float*)alloc(NE * 32 * 4), (float*)alloc(NE * 32 * 4)};
    float* sB[2]  = {(float*)alloc(NG * 16 * 4), (float*)alloc(NG * 16 * 4)};
    float* amc    = (float*)alloc(NN * DD * 4);
    float* agg    = (float*)alloc(NN * DD * 4);
    float* asum   = (float*)alloc(NG * 32 * 4);
    float* esum   = (float*)alloc(NG * 32 * 4);

    prep_kp<<<(DD * KPR + 255) / 256, 256, 0, stream>>>(kmat, bias, Kp);
    prep_wc<<<(128 * 160 + 255) / 256, 256, 0, stream>>>(We, Ue, Wc);

    for (int step = 0; step < 2; ++step) {
        const float* a_cur = step ? aB[0] : a0;
        const float* e_cur = step ? eB[0] : e0;
        const float* s_cur = step ? sB[0] : s0;
        float* a_nxt = aB[step];
        float* e_nxt = eB[step];
        float* s_nxt = sB[step];

        edge_gru_wmma<<<EG_BLOCKS, 256, 0, stream>>>(
            a_cur, e_cur, s_cur, pair, bgi, Wc, bei, ber, e_nxt);

        build_amc<<<(NN * DD + 255) / 256, 256, 0, stream>>>(a_cur, s_cur, agi, amc);
        hipMemsetAsync(agg, 0, NN * DD * 4, stream);

        msg_wmma<<<MSG_BLOCKS, 256, 0, stream>>>(e_nxt, amc, pair, Kp, agg);

        node_gru<<<(NN + 255) / 256, 256, 0, stream>>>(
            agg, a_cur, Wn, Un, bni, bnr, a_nxt);

        hipMemsetAsync(asum, 0, NG * 32 * 4, stream);
        hipMemsetAsync(esum, 0, NG * 32 * 4, stream);
        seg_sum32<<<(NN * 32 + 255) / 256, 256, 0, stream>>>(a_nxt, agi, asum, NN);
        seg_sum32<<<(NE * 32 + 255) / 256, 256, 0, stream>>>(e_nxt, bgi, esum, NE);

        state_gru<<<1, 128, 0, stream>>>(asum, esum, s_cur, Wsx, Usx, bsi, bsr, s_nxt);
    }

    float* out = (float*)d_out;
    hipMemcpyAsync(out,                     aB[1], NN * 32 * 4, hipMemcpyDeviceToDevice, stream);
    hipMemcpyAsync(out + NN * 32,           eB[1], NE * 32 * 4, hipMemcpyDeviceToDevice, stream);
    hipMemcpyAsync(out + NN * 32 + NE * 32, sB[1], NG * 16 * 4, hipMemcpyDeviceToDevice, stream);
}
